// CascadeRCNN_38328288150133
// MI455X (gfx1250) — matrix operand order repaired and checked by reading, not compiled, not run on hardware
//
#include <hip/hip_runtime.h>

// ---------- types for WMMA (CDNA5 gfx1250, wave32) ----------
typedef __bf16 bf16;
typedef bf16  v16bf __attribute__((ext_vector_type(16)));
typedef bf16  v8bf  __attribute__((ext_vector_type(8)));
typedef float v8f   __attribute__((ext_vector_type(8)));
typedef float v4f   __attribute__((ext_vector_type(4)));
typedef int   v4i   __attribute__((ext_vector_type(4)));

// global / LDS address-space pointer types for the async-copy builtin
typedef __attribute__((address_space(1))) v4i* gas_v4i_p;
typedef __attribute__((address_space(3))) v4i* las_v4i_p;

#if __has_builtin(__builtin_amdgcn_global_load_async_to_lds_b128) && \
    __has_builtin(__builtin_amdgcn_s_wait_asynccnt)
#define ASYNC_LDS 1
#else
#define ASYNC_LDS 0
#endif

static __device__ __forceinline__ unsigned short f32_to_bf16_rne(float f) {
    unsigned int u = __float_as_uint(f);
    unsigned int r = u + 0x7FFFu + ((u >> 16) & 1u);
    return (unsigned short)(r >> 16);
}
static __device__ __forceinline__ int imin(int a, int b) { return a < b ? a : b; }

// ======================================================================
// Kernel 0: streaming f32 -> bf16 conversion (one pass over each weight)
// ======================================================================
__global__ void __launch_bounds__(256)
cvt_f32_bf16_kernel(const float* __restrict__ src, unsigned short* __restrict__ dst, int n)
{
    const int i = (blockIdx.x * 256 + threadIdx.x) * 4;
    if (i < n) {
        const v4f v = *(const v4f*)(src + i);
        unsigned long long p =
              (unsigned long long)f32_to_bf16_rne(v[0])
            | ((unsigned long long)f32_to_bf16_rne(v[1]) << 16)
            | ((unsigned long long)f32_to_bf16_rne(v[2]) << 32)
            | ((unsigned long long)f32_to_bf16_rne(v[3]) << 48);
        *(unsigned long long*)(dst + i) = p;
    }
}

// ======================================================================
// Kernel 1: FPN level select + ROIAlignV2 (aligned=true, sampling_ratio=2)
// One block per ROI (1024 blocks), one thread per channel (256 threads).
// Writes pool5 as bf16, row-major [roi][c*49 + oy*7 + ox]  (K = 12544).
// ======================================================================
__global__ void __launch_bounds__(256)
roi_align_pool_kernel(const float* __restrict__ fm0, const float* __restrict__ fm1,
                      const float* __restrict__ fm2, const float* __restrict__ fm3,
                      const float* __restrict__ rois,
                      unsigned short* __restrict__ pool5)
{
    const int roi = blockIdx.x;
    const int c   = threadIdx.x;           // 0..255 channels

    const float* fms[4] = { fm0, fm1, fm2, fm3 };
    const int   Hs[4] = { 192, 96, 48, 24 };
    const int   Ws[4] = { 320, 160, 80, 40 };
    const float Ss[4] = { 4.f, 8.f, 16.f, 32.f };

    const float* r = rois + (size_t)roi * 5;
    const int   b  = (int)r[0];
    const float x1 = r[1], y1 = r[2], x2 = r[3], y2 = r[4];

    const float wh = fmaxf((x2 - x1) * (y2 - y1), 1e-6f);
    const float k  = floorf(4.0f + log2f(sqrtf(wh) / 224.0f));
    const int lvl  = (int)fminf(fmaxf(k, 2.0f), 5.0f) - 2;

    const int   H = Hs[lvl], W = Ws[lvl];
    const float s = Ss[lvl];
    const float* base = fms[lvl] + ((size_t)b * 256 + (size_t)c) * (size_t)H * (size_t)W;

    const float bx1 = x1 / s - 0.5f;
    const float by1 = y1 / s - 0.5f;
    const float dx  = (x2 - x1) / s;
    const float dy  = (y2 - y1) / s;

    unsigned short* out = pool5 + (size_t)roi * 12544 + (size_t)c * 49;

    for (int oy = 0; oy < 7; ++oy) {
        int   iy0[2], iy1[2];
        float wy0[2], wy1[2];
#pragma unroll
        for (int sy = 0; sy < 2; ++sy) {
            const float py = by1 + dy * ((float)(2 * oy + sy) + 0.5f) * (1.0f / 14.0f);
            const float valid = (py > -1.0f && py < (float)H) ? 1.0f : 0.0f;
            const float cc = fminf(fmaxf(py, 0.0f), (float)(H - 1));
            const int i0 = (int)floorf(cc);
            const int i1 = imin(i0 + 1, H - 1);
            const float fr = cc - (float)i0;
            iy0[sy] = i0; iy1[sy] = i1;
            wy0[sy] = (1.0f - fr) * valid;
            wy1[sy] = fr * valid;
        }
        for (int ox = 0; ox < 7; ++ox) {
            float acc = 0.0f;
#pragma unroll
            for (int sx = 0; sx < 2; ++sx) {
                const float px = bx1 + dx * ((float)(2 * ox + sx) + 0.5f) * (1.0f / 14.0f);
                const float valid = (px > -1.0f && px < (float)W) ? 1.0f : 0.0f;
                const float cc = fminf(fmaxf(px, 0.0f), (float)(W - 1));
                const int ix0 = (int)floorf(cc);
                const int ix1 = imin(ix0 + 1, W - 1);
                const float fr = cc - (float)ix0;
                const float wx0 = (1.0f - fr) * valid;
                const float wx1 = fr * valid;
#pragma unroll
                for (int sy = 0; sy < 2; ++sy) {
                    const float* r0 = base + (size_t)iy0[sy] * W;
                    const float* r1 = base + (size_t)iy1[sy] * W;
                    acc += wy0[sy] * (wx0 * r0[ix0] + wx1 * r0[ix1])
                         + wy1[sy] * (wx0 * r1[ix0] + wx1 * r1[ix1]);
                }
            }
            out[oy * 7 + ox] = f32_to_bf16_rne(acc * 0.25f);
        }
    }
}

// ======================================================================
// Kernel 2: LDS-tiled WMMA GEMM   C[M,N] = A[M,K] * W[N,K]^T + bias
// A, W both bf16. Block tile 64x128, BK=32, double-buffered LDS staged by
// gfx1250 async global->LDS loads (fallback: load+ds_store). 8 waves/block,
// each wave computes a 32x32 tile = 2x2 WMMA fragments (4 WMMA / K-step).
// Fragment layouts per CDNA5 ISA §7.12.2 (wave32):
//   A 16x32 : lane<16 -> M=lane, K∈{0..7,16..23}; lane>=16 -> M=lane-16, K∈{8..15,24..31}
//   B 32x16 : lane<16 -> N=lane, K∈{0..7,16..23}; lane>=16 -> N=lane-16, K∈{8..15,24..31}
//   C/D     : lane -> col N=lane%16; VGPR r -> row r + (lane<16?0:8)
// ======================================================================
#define BM 64
#define BN 128
#define BK 32
#define ATILE_B (BM * BK * 2)       // 4096 bytes
#define BTILE_B (BN * BK * 2)       // 8192 bytes
#define TILE_B  (ATILE_B + BTILE_B) // 12288 bytes

__global__ void __launch_bounds__(256)
gemm_wmma_lds_kernel(const unsigned short* __restrict__ A, int lda,
                     const unsigned short* __restrict__ Bw, int ldb,
                     const float* __restrict__ bias,
                     int Ntot, int K, int relu,
                     unsigned short* __restrict__ outB,
                     float* __restrict__ outF, int ldo,
                     int nBlocksN)
{
    __shared__ __align__(16) unsigned char smem[2 * TILE_B + 256]; // +dummy sink

    const int t    = threadIdx.x;
    const int wave = t >> 5;
    const int lane = t & 31;
    const int hi   = (lane >> 4) & 1;
    const int l16  = lane & 15;

    const int tm0 = (blockIdx.x / nBlocksN) * BM;
    const int tn0 = (blockIdx.x % nBlocksN) * BN;

    // loader assignment: A: 64 rows x 4 x 16B; B: 128 rows x 2 x 32B
    const int arow = t >> 2, aseg = t & 3;    // 16B seg of a 64B row
    const int brow = t >> 1, bseg = t & 1;    // 32B half of a 64B row
    const int gn       = tn0 + brow;
    const bool brow_ok = (gn < Ntot);
    const int gn_c     = brow_ok ? gn : 0;

    // zero-fill B rows that fall outside Ntot (both buffers), once
    if (Ntot < tn0 + BN) {
        for (int task = t; task < 2 * BN * 4; task += 256) {
            const int buf = task >> 9;
            const int row = (task & 511) >> 2;
            const int seg = task & 3;
            if (tn0 + row >= Ntot) {
                v4i z = {};
                *(v4i*)(smem + buf * TILE_B + ATILE_B + row * 64 + seg * 16) = z;
            }
        }
    }
    __syncthreads();

    const int nk = K / BK;
    const int wm = (wave >> 2) * 32;   // 0 / 32
    const int wn = (wave & 3) * 32;    // 0 / 32 / 64 / 96

    v8f acc[2][2] = {};

    auto issueTile = [&](int buf, int k0) {
        unsigned char* Ab = smem + buf * TILE_B;
        unsigned char* Bb = Ab + ATILE_B;
        const unsigned short* ga = A  + (size_t)(tm0 + arow) * (size_t)lda + k0 + aseg * 8;
        const unsigned short* gb = Bw + (size_t)gn_c * (size_t)ldb + k0 + bseg * 16;
        unsigned char* la  = Ab + arow * 64 + aseg * 16;
        unsigned char* lb0 = brow_ok ? (Bb + brow * 64 + bseg * 32) : (smem + 2 * TILE_B);
        unsigned char* lb1 = brow_ok ? (lb0 + 16) : (smem + 2 * TILE_B);
#if ASYNC_LDS
        __builtin_amdgcn_global_load_async_to_lds_b128(
            (gas_v4i_p)ga, (las_v4i_p)la, 0, 0);
        __builtin_amdgcn_global_load_async_to_lds_b128(
            (gas_v4i_p)gb, (las_v4i_p)lb0, 0, 0);
        __builtin_amdgcn_global_load_async_to_lds_b128(
            (gas_v4i_p)(gb + 8), (las_v4i_p)lb1, 0, 0);
#else
        *(v4i*)la  = *(const v4i*)ga;
        *(v4i*)lb0 = *(const v4i*)gb;
        *(v4i*)lb1 = *(const v4i*)(gb + 8);
#endif
    };

    issueTile(0, 0);

    for (int i = 0; i < nk; ++i) {
        const bool more = (i + 1 < nk);
        if (more) issueTile((i + 1) & 1, (i + 1) * BK);
#if ASYNC_LDS
        if (more) __builtin_amdgcn_s_wait_asynccnt(3);
        else      __builtin_amdgcn_s_wait_asynccnt(0);
#endif
        __syncthreads();

        // far-ahead prefetch of the weight stream into L2
        if (i + 2 < nk)
            __builtin_prefetch((const void*)(Bw + (size_t)gn_c * (size_t)ldb
                                             + (i + 2) * BK + bseg * 16), 0, 1);

        const unsigned short* Ab = (const unsigned short*)(smem + (i & 1) * TILE_B);
        const unsigned short* Bb = Ab + BM * BK;

        v16bf af[2], bfr[2];
#pragma unroll
        for (int mi = 0; mi < 2; ++mi) {
            const unsigned short* p = Ab + (wm + mi * 16 + l16) * BK + (hi ? 8 : 0);
            const v8bf x0 = *(const v8bf*)p;
            const v8bf x1 = *(const v8bf*)(p + 16);
#pragma unroll
            for (int e = 0; e < 8; ++e) { af[mi][e] = x0[e]; af[mi][8 + e] = x1[e]; }
        }
#pragma unroll
        for (int ni = 0; ni < 2; ++ni) {
            const unsigned short* q = Bb + (wn + ni * 16 + l16) * BK + (hi ? 8 : 0);
            const v8bf y0 = *(const v8bf*)q;
            const v8bf y1 = *(const v8bf*)(q + 16);
#pragma unroll
            for (int e = 0; e < 8; ++e) { bfr[ni][e] = y0[e]; bfr[ni][8 + e] = y1[e]; }
        }

#pragma unroll
        for (int mi = 0; mi < 2; ++mi)
#pragma unroll
            for (int ni = 0; ni < 2; ++ni)
                acc[mi][ni] = __builtin_amdgcn_wmma_f32_16x16x32_bf16(
                    false, af[mi], false, bfr[ni], (short)0, acc[mi][ni], false, false);

        __syncthreads();
    }

    // epilogue: bias + optional relu, scatter to bf16 and/or f32 outputs
#pragma unroll
    for (int ni = 0; ni < 2; ++ni) {
        const int n = tn0 + wn + ni * 16 + l16;
        if (n >= Ntot) continue;
        const float bv = bias[n];
#pragma unroll
        for (int mi = 0; mi < 2; ++mi) {
#pragma unroll
            for (int r = 0; r < 8; ++r) {
                float v = acc[mi][ni][r] + bv;
                if (relu) v = fmaxf(v, 0.0f);
                const int m = tm0 + wm + mi * 16 + r + hi * 8;
                if (outB) outB[(size_t)m * (size_t)ldo + n] = f32_to_bf16_rne(v);
                if (outF) outF[(size_t)m * (size_t)ldo + n] = v;
            }
        }
    }
}

// ======================================================================
// Launch: cvt weights -> ROIAlign -> fc1(relu) -> fc2(relu) -> proj(f32)
// ======================================================================
extern "C" void kernel_launch(void* const* d_in, const int* in_sizes, int n_in,
                              void* d_out, int out_size, void* d_ws, size_t ws_size,
                              hipStream_t stream)
{
    const float* fm0   = (const float*)d_in[0];
    const float* fm1   = (const float*)d_in[1];
    const float* fm2   = (const float*)d_in[2];
    const float* fm3   = (const float*)d_in[3];
    const float* rois  = (const float*)d_in[4];
    const float* fc1_w = (const float*)d_in[5];
    const float* fc1_b = (const float*)d_in[6];
    const float* fc2_w = (const float*)d_in[7];
    const float* fc2_b = (const float*)d_in[8];
    const float* p_w   = (const float*)d_in[9];
    const float* p_b   = (const float*)d_in[10];

    const int N = 1024, D = 12544, F = 1024, NO = 20;

    unsigned short* pool5 = (unsigned short*)d_ws;                  // [1024,12544] bf16
    unsigned short* act1  = pool5 + (size_t)N * D;                  // [1024,1024]
    unsigned short* act2  = act1  + (size_t)N * F;                  // [1024,1024]
    unsigned short* w1    = act2  + (size_t)N * F;                  // [1024,12544]
    unsigned short* w2    = w1    + (size_t)F * D;                  // [1024,1024]
    unsigned short* pw    = w2    + (size_t)F * F;                  // [20,1024]

    // 0) one-pass weight conversion f32 -> bf16
    cvt_f32_bf16_kernel<<<(F * D) / 4 / 256, 256, 0, stream>>>(fc1_w, w1, F * D);
    cvt_f32_bf16_kernel<<<(F * F) / 4 / 256, 256, 0, stream>>>(fc2_w, w2, F * F);
    cvt_f32_bf16_kernel<<<(NO * F) / 4 / 256, 256, 0, stream>>>(p_w, pw, NO * F);

    // 1) ROIAlign + level select -> bf16 pool5
    roi_align_pool_kernel<<<N, 256, 0, stream>>>(fm0, fm1, fm2, fm3, rois, pool5);

    // 2) fc1: M=1024, N=1024, K=12544 -> (1024/64)*(1024/128) = 128 blocks
    gemm_wmma_lds_kernel<<<(N / BM) * (F / BN), 256, 0, stream>>>(
        pool5, D, w1, D, fc1_b, F, D, /*relu=*/1, act1, nullptr, F, F / BN);

    // 3) fc2: M=1024, N=1024, K=1024
    gemm_wmma_lds_kernel<<<(N / BM) * (F / BN), 256, 0, stream>>>(
        act1, F, w2, F, fc2_b, F, F, /*relu=*/1, act2, nullptr, F, F / BN);

    // 4) proj: M=1024, N=20 (one 128-wide N block), K=1024 -> f32 out [1024,20]
    gemm_wmma_lds_kernel<<<(N / BM) * 1, 256, 0, stream>>>(
        act2, F, pw, F, p_b, NO, F, /*relu=*/0, nullptr, (float*)d_out, NO, 1);
}
